// MultiHeadAttention__46462956208368
// MI455X (gfx1250) — matrix operand order repaired and checked by reading, not compiled, not run on hardware
//
#include <hip/hip_runtime.h>

// MHA forward for B=4, S=4096, D=512, H=2, PD=256 on gfx1250 (wave32, WMMA).
// Pipeline: W fp32->bf16 | QKV GEMMs (bf16 WMMA, f32 accum, Q pre-scaled by
// 1/sqrt(PD)) | fused flash-attention (bf16 WMMA QK^T and PV, f32 online
// softmax, TDM double-buffered K tiles) | output projection GEMM -> fp32.
// Workspace layout (needs ~66 MB): 4x bf16 weights (2MB) + Q,K,V,attnO bf16.

typedef unsigned short u16;
typedef unsigned int   u32;
typedef unsigned long long u64;

typedef __attribute__((ext_vector_type(16))) __bf16 bf16x16;
typedef __attribute__((ext_vector_type(8)))  float  f32x8;
typedef __attribute__((ext_vector_type(4)))  unsigned int u32x4;
typedef __attribute__((ext_vector_type(4)))  int          i32x4;
typedef __attribute__((ext_vector_type(8)))  int          i32x8;

__device__ __forceinline__ u16 f2bf(float f) {
  u32 u = __float_as_uint(f);
  u32 r = u + 0x7FFFu + ((u >> 16) & 1u);   // round-to-nearest-even
  return (u16)(r >> 16);
}

union FragU { uint4 q[2]; bf16x16 v; };
union AccU  { f32x8 v; float f[8]; };

// Load one 16x32 bf16 WMMA A-frag (or 32x16 B-frag) from row-major storage.
// Lane L holds row/col (row_base + L&15); its 16 bf16 are two contiguous
// 8-element runs at k = kbase + 8*(L>>4) and k = kbase + 16 + 8*(L>>4).
__device__ __forceinline__ bf16x16 load_frag_rm(const u16* base, int ld,
                                                int row_base, int kbase) {
  const int lane = threadIdx.x & 31;
  const int r    = row_base + (lane & 15);
  const int hh   = lane >> 4;
  const u16* p = base + (size_t)r * ld + kbase + hh * 8;
  FragU f;
  f.q[0] = *(const uint4*)(p);
  f.q[1] = *(const uint4*)(p + 16);
  return f.v;
}

__device__ __forceinline__ float redmax16(float v) {
  v = fmaxf(v, __shfl_xor(v, 1, 32));
  v = fmaxf(v, __shfl_xor(v, 2, 32));
  v = fmaxf(v, __shfl_xor(v, 4, 32));
  v = fmaxf(v, __shfl_xor(v, 8, 32));
  return v;
}
__device__ __forceinline__ float redsum16(float v) {
  v += __shfl_xor(v, 1, 32);
  v += __shfl_xor(v, 2, 32);
  v += __shfl_xor(v, 4, 32);
  v += __shfl_xor(v, 8, 32);
  return v;
}

// Issue a TDM 2D tile load: bf16 tile [64 rows x 256 cols], row stride 512
// elements, into LDS (row-major, ld 256). Built per CDNA5 D# layout
// (group0: count/lds/global/type, group1: data_size=2B, dims/strides).
__device__ __forceinline__ void tdm_load_k_tile(const u16* lds_ptr,
                                                const u16* gptr) {
  const u32 lds_off = (u32)(u64)lds_ptr;            // flat->LDS low 32 bits
  const u64 ga = (u64)gptr;
  u32x4 g0;
  g0.x = 1u;                                        // count=1, user D#
  g0.y = lds_off;                                   // lds_addr (bytes)
  g0.z = (u32)ga;                                   // global_addr[31:0]
  g0.w = (u32)(ga >> 32) | (2u << 30);              // [56:32] | type=2
  i32x8 g1;
  g1[0] = (int)(1u << 16);                          // data_size=1 (2 bytes)
  g1[1] = (int)(256u << 16);                        // tensor_dim0[15:0]
  g1[2] = (int)(64u << 16);                         // d0 hi=0 | tensor_dim1 lo
  g1[3] = (int)(256u << 16);                        // d1 hi=0 | tile_dim0=256
  g1[4] = 64;                                       // tile_dim1=64, tile_dim2=0
  g1[5] = 512;                                      // tensor_dim0_stride
  g1[6] = 0;
  g1[7] = 0;
  i32x4 gz4 = {0, 0, 0, 0};
  i32x8 gz8 = {0, 0, 0, 0, 0, 0, 0, 0};
  __builtin_amdgcn_tensor_load_to_lds(g0, g1, gz4, gz4, gz8, 0);
}

// ---------------------------------------------------------------- wcvt ----
__global__ void wcvt_kernel(const float* __restrict__ src, u16* __restrict__ dst,
                            int n) {
  int i = blockIdx.x * blockDim.x + threadIdx.x;
  if (i < n) dst[i] = f2bf(src[i]);
}

// ------------------------------------------------------- GEMM (N=K=512) ----
// Out[M,512] = (A[M,512] @ W[512,512] + bias) * scale
// Block: 256 threads (8 waves), tile 128(M) x 128(N); wave = 32x64 subtile.
template<bool A_F32, bool OUT_BF16>
__global__ __launch_bounds__(256) void gemm_bias_512(
    const void* __restrict__ Aptr, const u16* __restrict__ W,
    const float* __restrict__ bias, void* __restrict__ Out, float scale) {
  __shared__ u16 xs [128 * 32];   // A tile, row-major [m][k], ld 32
  __shared__ u16 wst[128 * 32];   // W tile transposed [n][k], ld 32

  const int tid  = threadIdx.x;
  const int wave = tid >> 5;
  const int lane = tid & 31;
  const int wm   = wave & 3;      // 4 waves along M
  const int wn   = wave >> 2;     // 2 waves along N
  const int mblk = blockIdx.x * 128;
  const int nblk = blockIdx.y * 128;

  AccU acc[2][4];
#pragma unroll
  for (int i = 0; i < 2; ++i)
#pragma unroll
    for (int j = 0; j < 4; ++j)
#pragma unroll
      for (int v = 0; v < 8; ++v) acc[i][j].f[v] = 0.f;

  for (int kb = 0; kb < 512; kb += 32) {
    // stage A tile (convert fp32->bf16 if needed)
    {
      const int r  = tid >> 1;
      const int kc = (tid & 1) * 16;
      union { u16 h[16]; uint4 q[2]; } tmp;
      if (A_F32) {
        const float* src = (const float*)Aptr + (size_t)(mblk + r) * 512 + kb + kc;
#pragma unroll
        for (int e = 0; e < 16; ++e) tmp.h[e] = f2bf(src[e]);
      } else {
        const u16* src = (const u16*)Aptr + (size_t)(mblk + r) * 512 + kb + kc;
        tmp.q[0] = *(const uint4*)(src);
        tmp.q[1] = *(const uint4*)(src + 8);
      }
      *(uint4*)&xs[r * 32 + kc]     = tmp.q[0];
      *(uint4*)&xs[r * 32 + kc + 8] = tmp.q[1];
    }
    // stage W tile transposed (coalesced read, scattered b16 LDS writes)
    {
      const int k  = tid >> 3;
      const int c0 = (tid & 7) * 16;
      union { u16 h[16]; uint4 q[2]; } tmp;
      const u16* src = W + (size_t)(kb + k) * 512 + nblk + c0;
      tmp.q[0] = *(const uint4*)(src);
      tmp.q[1] = *(const uint4*)(src + 8);
#pragma unroll
      for (int e = 0; e < 16; ++e) wst[(c0 + e) * 32 + k] = tmp.h[e];
    }
    __syncthreads();

    bf16x16 af[2], bfr[4];
#pragma unroll
    for (int i = 0; i < 2; ++i) af[i]  = load_frag_rm(xs,  32, wm * 32 + i * 16, 0);
#pragma unroll
    for (int j = 0; j < 4; ++j) bfr[j] = load_frag_rm(wst, 32, wn * 64 + j * 16, 0);
#pragma unroll
    for (int i = 0; i < 2; ++i)
#pragma unroll
      for (int j = 0; j < 4; ++j)
        acc[i][j].v = __builtin_amdgcn_wmma_f32_16x16x32_bf16(
            false, af[i], false, bfr[j], (short)0, acc[i][j].v, false, false);
    __syncthreads();
  }

  const int nloc = lane & 15;
  const int hh   = lane >> 4;
#pragma unroll
  for (int i = 0; i < 2; ++i)
#pragma unroll
    for (int j = 0; j < 4; ++j) {
      const int col = nblk + wn * 64 + j * 16 + nloc;
      const float bcol = bias[col];
#pragma unroll
      for (int v = 0; v < 8; ++v) {
        const int row = mblk + wm * 32 + i * 16 + v + 8 * hh;
        const float val = (acc[i][j].f[v] + bcol) * scale;
        if (OUT_BF16) ((u16*) Out)[(size_t)row * 512 + col] = f2bf(val);
        else          ((float*)Out)[(size_t)row * 512 + col] = val;
      }
    }
}

// ----------------------------------------------------- flash attention ----
// grid (S/128, B*H); block 256 (8 waves). Wave owns 16 query rows, full
// PD=256 accumulator. Key tiles of 64; K tiles are TDM double-buffered
// (wave 0 issues tensor_load_to_lds for tile kt+1 while all compute kt).
__global__ __launch_bounds__(256) void flash_attn(
    const u16* __restrict__ Qb, const u16* __restrict__ Kb,
    const u16* __restrict__ Vb, u16* __restrict__ Ob) {
  extern __shared__ char smem[];
  u16* Kb0 = (u16*)smem;           // K tile ping: 64 keys x 256 d, ld 256
  u16* Kb1 = Kb0 + 64 * 256;       // K tile pong
  u16* Vt  = Kb1 + 64 * 256;       // transposed V: 256 d x (ld 72) keys
  u16* Pb  = Vt + 256 * 72;        // per-wave P: 8 x [16 x 64]

  const int tid  = threadIdx.x;
  const int wave = tid >> 5;
  const int lane = tid & 31;
  const int nloc = lane & 15;
  const int hh   = lane >> 4;
  const int b    = blockIdx.y >> 1;
  const int h    = blockIdx.y & 1;
  const size_t rowbase = (size_t)b * 4096;
  const int q0 = blockIdx.x * 128 + wave * 16;
  const u16* Khead = Kb + rowbase * 512 + h * 256;   // head slice, ld 512

  // wave-0-only scalar branch for TDM issue (readfirstlane -> s_cbranch)
  const bool issuer = (__builtin_amdgcn_readfirstlane(tid) == 0);
  if (issuer) tdm_load_k_tile(Kb0, Khead);           // prologue: tile 0

  // resident Q A-frags (16 rows x 256 d = 8 k-steps)
  const u16* Qrow = Qb + (rowbase + q0) * 512 + h * 256;
  bf16x16 qf[8];
#pragma unroll
  for (int kd = 0; kd < 8; ++kd) qf[kd] = load_frag_rm(Qrow, 512, 0, kd * 32);

  AccU of[16];
#pragma unroll
  for (int n = 0; n < 16; ++n)
#pragma unroll
    for (int v = 0; v < 8; ++v) of[n].f[v] = 0.f;
  float mrow[8], lrow[8];
#pragma unroll
  for (int v = 0; v < 8; ++v) { mrow[v] = -3.0e38f; lrow[v] = 0.f; }

  u16* Pw = Pb + wave * 16 * 64;

  for (int kt = 0; kt < 64; ++kt) {
    const int key0 = kt * 64;
    // issue TDM for next K tile, then wait for current tile (in-order TDM)
    if (issuer) {
      if (kt + 1 < 64) {
        tdm_load_k_tile((kt & 1) ? Kb0 : Kb1,
                        Khead + (size_t)(key0 + 64) * 512);
        __builtin_amdgcn_s_wait_tensorcnt(1);
      } else {
        __builtin_amdgcn_s_wait_tensorcnt(0);
      }
    }
    // cooperative stage: V transposed into LDS; prefetch next V tile
    {
      const int r  = tid >> 2;          // key row 0..63
      const int c0 = (tid & 3) * 64;    // d chunk
      const u16* vsrc = Vb + (rowbase + key0 + r) * 512 + h * 256 + c0;
      if (kt + 1 < 64)
        __builtin_prefetch((const void*)(vsrc + (size_t)64 * 512), 0, 1);
#pragma unroll
      for (int u = 0; u < 8; ++u) {
        union { u16 e[8]; uint4 q; } t;
        t.q = *(const uint4*)(vsrc + u * 8);
#pragma unroll
        for (int jj = 0; jj < 8; ++jj) Vt[(c0 + u * 8 + jj) * 72 + r] = t.e[jj];
      }
    }
    __syncthreads();
    const u16* Kcur = (kt & 1) ? Kb1 : Kb0;

    // S = Q K^T  (16 x 64, Q pre-scaled by 1/sqrt(PD))
    AccU s[4];
#pragma unroll
    for (int j = 0; j < 4; ++j) {
#pragma unroll
      for (int v = 0; v < 8; ++v) s[j].f[v] = 0.f;
#pragma unroll
      for (int kd = 0; kd < 8; ++kd) {
        bf16x16 kfrag = load_frag_rm(Kcur, 256, j * 16, kd * 32);
        s[j].v = __builtin_amdgcn_wmma_f32_16x16x32_bf16(
            false, qf[kd], false, kfrag, (short)0, s[j].v, false, false);
      }
    }

    // online softmax (row stats via 16-lane shfl reductions)
    float alpha[8];
#pragma unroll
    for (int v = 0; v < 8; ++v) {
      float rmax = s[0].f[v];
#pragma unroll
      for (int j = 1; j < 4; ++j) rmax = fmaxf(rmax, s[j].f[v]);
      rmax = redmax16(rmax);
      const float mnew = fmaxf(mrow[v], rmax);
      alpha[v] = __expf(mrow[v] - mnew);
      float rsum = 0.f;
#pragma unroll
      for (int j = 0; j < 4; ++j) {
        const float p = __expf(s[j].f[v] - mnew);
        s[j].f[v] = p;
        rsum += p;
      }
      rsum = redsum16(rsum);
      lrow[v] = lrow[v] * alpha[v] + rsum;
      mrow[v] = mnew;
    }
#pragma unroll
    for (int n = 0; n < 16; ++n)
#pragma unroll
      for (int v = 0; v < 8; ++v) of[n].f[v] *= alpha[v];

    // P (C-layout) -> wave-private LDS -> A-layout frags
#pragma unroll
    for (int j = 0; j < 4; ++j)
#pragma unroll
      for (int v = 0; v < 8; ++v)
        Pw[(v + 8 * hh) * 64 + j * 16 + nloc] = f2bf(s[j].f[v]);
    __syncthreads();

    bf16x16 pa[2];
#pragma unroll
    for (int kk = 0; kk < 2; ++kk) pa[kk] = load_frag_rm(Pw, 64, 0, kk * 32);

    // O += P V   (B-frags from transposed V tile)
#pragma unroll
    for (int n = 0; n < 16; ++n)
#pragma unroll
      for (int kk = 0; kk < 2; ++kk) {
        bf16x16 vfrag = load_frag_rm(Vt, 72, n * 16, kk * 32);
        of[n].v = __builtin_amdgcn_wmma_f32_16x16x32_bf16(
            false, pa[kk], false, vfrag, (short)0, of[n].v, false, false);
      }
    __syncthreads();
  }

  float linv[8];
#pragma unroll
  for (int v = 0; v < 8; ++v) linv[v] = 1.f / lrow[v];
#pragma unroll
  for (int n = 0; n < 16; ++n)
#pragma unroll
    for (int v = 0; v < 8; ++v) {
      const size_t row = rowbase + q0 + v + 8 * hh;
      const int col = h * 256 + n * 16 + nloc;
      Ob[row * 512 + col] = f2bf(of[n].f[v] * linv[v]);
    }
}

// ------------------------------------------------------------ launcher ----
extern "C" void kernel_launch(void* const* d_in, const int* in_sizes, int n_in,
                              void* d_out, int out_size, void* d_ws, size_t ws_size,
                              hipStream_t stream) {
  (void)in_sizes; (void)n_in; (void)out_size; (void)ws_size;
  const float* x  = (const float*)d_in[0];
  const float* Wq = (const float*)d_in[1];
  const float* bq = (const float*)d_in[2];
  const float* Wk = (const float*)d_in[3];
  const float* bk = (const float*)d_in[4];
  const float* Wv = (const float*)d_in[5];
  const float* bv = (const float*)d_in[6];
  const float* Wo = (const float*)d_in[7];
  const float* bo = (const float*)d_in[8];

  const size_t WBYTES = (size_t)512 * 512 * 2;        // one bf16 weight
  const size_t QBYTES = (size_t)16384 * 512 * 2;      // one bf16 activation
  char* ws = (char*)d_ws;
  u16* Wqb = (u16*)(ws + 0 * WBYTES);
  u16* Wkb = (u16*)(ws + 1 * WBYTES);
  u16* Wvb = (u16*)(ws + 2 * WBYTES);
  u16* Wob = (u16*)(ws + 3 * WBYTES);
  u16* Qbf = (u16*)(ws + 4 * WBYTES);
  u16* Kbf = (u16*)(ws + 4 * WBYTES + 1 * QBYTES);
  u16* Vbf = (u16*)(ws + 4 * WBYTES + 2 * QBYTES);
  u16* Obf = (u16*)(ws + 4 * WBYTES + 3 * QBYTES);

  const int WN = 512 * 512;
  wcvt_kernel<<<WN / 256, 256, 0, stream>>>(Wq, Wqb, WN);
  wcvt_kernel<<<WN / 256, 256, 0, stream>>>(Wk, Wkb, WN);
  wcvt_kernel<<<WN / 256, 256, 0, stream>>>(Wv, Wvb, WN);
  wcvt_kernel<<<WN / 256, 256, 0, stream>>>(Wo, Wob, WN);

  dim3 gg(16384 / 128, 512 / 128);
  gemm_bias_512<true, true><<<gg, 256, 0, stream>>>(x, Wqb, bq, Qbf, 1.0f / 16.0f);
  gemm_bias_512<true, true><<<gg, 256, 0, stream>>>(x, Wkb, bk, Kbf, 1.0f);
  gemm_bias_512<true, true><<<gg, 256, 0, stream>>>(x, Wvb, bv, Vbf, 1.0f);

  const size_t smem =
      (size_t)(2 * 64 * 256 + 256 * 72 + 8 * 16 * 64) * sizeof(u16);
  flash_attn<<<dim3(4096 / 128, 4 * 2), 256, smem, stream>>>(Qbf, Kbf, Vbf, Obf);

  gemm_bias_512<false, false><<<gg, 256, 0, stream>>>(Obf, Wob, bo, d_out, 1.0f);
}
